// rnn_cnn_layer_88416196756028
// MI455X (gfx1250) — matrix operand order repaired and checked by reading, not compiled, not run on hardware
//
#include <hip/hip_runtime.h>

// ---------------------------------------------------------------------------
// Shapes
// ---------------------------------------------------------------------------
#define B_   64
#define T_   1024
#define F_   64
#define H_   256
#define G4   1024            // 4*H
#define KCAT 320             // F_ + H_   (K of the recurrent GEMM)
#define KA   328             // padded LDS row stride in halves (mult of 8)
#define TO   1020            // T - K + 1
#define KCNV 1280            // 5 * H

typedef _Float16 v16h  __attribute__((ext_vector_type(16)));
typedef _Float16 f16x8 __attribute__((ext_vector_type(8)));
typedef _Float16 f16x4 __attribute__((ext_vector_type(4)));
typedef float    v8f   __attribute__((ext_vector_type(8)));

union Frag { v16h v; f16x8 h[2]; };

__device__ __forceinline__ float sigf(float x) { return 1.0f / (1.0f + __expf(-x)); }

// ---------------------------------------------------------------------------
// Pack [W;U] (f32, [K][4H]) -> WcatT (f16, [N=1024][K=320], K contiguous).
// Per-lane B fragments become two aligned global_load_b128.
// ---------------------------------------------------------------------------
__global__ __launch_bounds__(256)
void pack_wcat(const float* __restrict__ W, const float* __restrict__ U,
               _Float16* __restrict__ WcatT) {
  const int n = blockIdx.x;                       // 0..1023
  for (int k = threadIdx.x; k < KCAT; k += 256) {
    float v = (k < F_) ? W[k * G4 + n] : U[(k - F_) * G4 + n];
    WcatT[(size_t)n * KCAT + k] = (_Float16)v;
  }
}

// Pack Wc (f32, [5][H][256]) -> WcT (f16, [N=256][K=1280], K contiguous).
__global__ __launch_bounds__(256)
void pack_wc(const float* __restrict__ Wc, _Float16* __restrict__ WcT) {
  const int n = blockIdx.x;                       // 0..255
  for (int kk = threadIdx.x; kk < KCNV; kk += 256) {
    const int tap = kk >> 8, c = kk & 255;
    WcT[(size_t)n * KCNV + kk] = (_Float16)Wc[((size_t)tap * H_ + c) * 256 + n];
  }
}

// ---------------------------------------------------------------------------
// LSTM recurrence: 16-workgroup cluster, one WGP per 16-neuron group.
//   A = [x_t | h_{t-1}]  (64 x 320, f16, staged in LDS)
//   B = WcatT columns for this WG's 4 gate tiles (compiler register-hoists
//       all 40 fragments out of the time loop -> U is register-resident)
//   Gates/cell-state update are pure register math (C/D layout alignment).
//   h exchanged via time-indexed L2 buffer + cluster barrier.
// ---------------------------------------------------------------------------
__global__ __launch_bounds__(128)
void lstm_recur(const float* __restrict__ x,      // [B][T][F]
                const float* __restrict__ bias,   // [4H]
                const _Float16* __restrict__ WcatT,
                _Float16* __restrict__ hbuf) {    // [T][B][H] f16
  __shared__ _Float16 a_buf[B_ * KA];

  const int tid  = threadIdx.x;
  const int lane = tid & 31;
  const int wave = tid >> 5;        // 0..3  -> M tile
  const int lh   = lane >> 4;       // lane half
  const int lm   = lane & 15;
  const int n0   = blockIdx.x * 16; // neuron group base
  const int mt   = wave;

  const float bI = bias[0 * H_ + n0 + lm];
  const float bF = bias[1 * H_ + n0 + lm];
  const float bG = bias[2 * H_ + n0 + lm];
  const float bO = bias[3 * H_ + n0 + lm];

  float cst[8];
#pragma unroll
  for (int r = 0; r < 8; ++r) cst[r] = 0.0f;

  for (int t = 0; t < T_; ++t) {
    // ---- stage A = [x_t | h_{t-1}] into LDS (all lanes uniform trip counts)
    // x_t: 1024 float4 loads -> packed f16x4 stores (8 iters/thread)
    {
      const float4* xsrc = reinterpret_cast<const float4*>(x + (size_t)t * F_);
      for (int i = tid; i < (B_ * F_) / 4; i += 128) {   // 8 iters
        const int b = i >> 4, f4 = i & 15;               // 16 float4 per row
        const float4 v = xsrc[(size_t)b * (T_ * F_ / 4) + f4];
        f16x4 p = { (_Float16)v.x, (_Float16)v.y, (_Float16)v.z, (_Float16)v.w };
        *reinterpret_cast<f16x4*>(&a_buf[b * KA + f4 * 4]) = p;
      }
    }
    if (t == 0) {
      for (int i = tid; i < B_ * H_; i += 128) {        // 128 iters
        const int b = i >> 8, c = i & 255;
        a_buf[b * KA + F_ + c] = (_Float16)0.0f;
      }
    } else {
      const f16x8* src =
          reinterpret_cast<const f16x8*>(hbuf + (size_t)(t - 1) * B_ * H_);
      for (int i = tid; i < (B_ * H_) / 8; i += 128) {  // 16 iters, b128
        const int b = i >> 5, c8 = i & 31;
        *reinterpret_cast<f16x8*>(&a_buf[b * KA + F_ + c8 * 8]) = src[i];
      }
    }
    __syncthreads();

    // ---- gather A fragments once, reuse across all 4 gates
    Frag afr[10];
#pragma unroll
    for (int kt = 0; kt < 10; ++kt) {
      const _Float16* row = &a_buf[(mt * 16 + lm) * KA + kt * 32];
      afr[kt].h[0] = *reinterpret_cast<const f16x8*>(row + lh * 8);
      afr[kt].h[1] = *reinterpret_cast<const f16x8*>(row + 16 + lh * 8);
    }

    // ---- four gate GEMM tiles: K = 320 -> 10 WMMAs each
    v8f acc[4];
#pragma unroll
    for (int g = 0; g < 4; ++g) {
      const float bv = (g == 0) ? bI : (g == 1) ? bF : (g == 2) ? bG : bO;
      v8f a8 = {bv, bv, bv, bv, bv, bv, bv, bv};
      const _Float16* col =
          WcatT + (size_t)(g * H_ + n0 + lm) * KCAT + lh * 8;
#pragma unroll
      for (int kt = 0; kt < 10; ++kt) {
        Frag bfr;
        bfr.h[0] = *reinterpret_cast<const f16x8*>(col + kt * 32);
        bfr.h[1] = *reinterpret_cast<const f16x8*>(col + kt * 32 + 16);
        a8 = __builtin_amdgcn_wmma_f32_16x16x32_f16(false, afr[kt].v,
                                                    false, bfr.v,
                                                    (short)0, a8, false, false);
      }
      acc[g] = a8;
    }

    // ---- gates + cell state + h (register-local: same lane/reg across gates)
#pragma unroll
    for (int r = 0; r < 8; ++r) {
      const float ig = sigf(acc[0][r]);
      const float fg = sigf(acc[1][r]);
      const float zg = acc[2][r];
      const float gg = zg * sigf(zg);               // silu candidate
      const float og = sigf(acc[3][r]);
      const float c  = fg * cst[r] + ig * gg;
      cst[r] = c;
      const float h = og * (c * sigf(c));           // silu(cell)
      const int brow = mt * 16 + r + 8 * lh;        // C/D layout row
      hbuf[((size_t)t * B_ + brow) * H_ + n0 + lm] = (_Float16)h;
    }

    __syncthreads();       // a_buf free for next step's staging
    __threadfence();       // publish hbuf[t] to L2 for the other 15 WGPs
#if __has_builtin(__builtin_amdgcn_s_cluster_barrier)
    __builtin_amdgcn_s_cluster_barrier();   // s_barrier_signal/wait -3
#endif
  }
}

// ---------------------------------------------------------------------------
// Conv1D (VALID, K=5) as im2col GEMM + fused bias/SiLU.
// Each wave computes a 16x64 output tile (4 N-tiles) so the A fragment is
// reused across 4 WMMAs: 10 b128 loads per 4 WMMAs instead of 16.
// K = 1280 -> 40 k-steps -> 160 f16 WMMAs per wave.
// ---------------------------------------------------------------------------
__global__ __launch_bounds__(128)
void conv_silu(const _Float16* __restrict__ hbuf,  // [T][B][H]
               const _Float16* __restrict__ WcT,   // [256][1280]
               const float* __restrict__ bc,       // [256]
               float* __restrict__ out) {          // [B][TO][256]
  const int tid   = threadIdx.x;
  const int lane  = tid & 31;
  const int wave  = tid >> 5;                      // 0..3
  const int lh    = lane >> 4;
  const int lm    = lane & 15;
  const int mtile = blockIdx.x;                    // 0..4079
  const int n0    = wave * 64;                     // 4 N-tiles per wave

  // A-layout row for this lane (lanes 0-15 and 16-31 both cover rows 0-15)
  const int row = mtile * 16 + lm;
  const int b   = row / TO;
  const int t   = row % TO;

  v8f acc[4];
  const _Float16* colB[4];
#pragma unroll
  for (int j = 0; j < 4; ++j) {
    const float bv = bc[n0 + j * 16 + lm];
    acc[j] = (v8f){bv, bv, bv, bv, bv, bv, bv, bv};
    colB[j] = WcT + (size_t)(n0 + j * 16 + lm) * KCNV + lh * 8;
  }

#pragma unroll 4
  for (int kt = 0; kt < 40; ++kt) {
    const int tap = kt >> 3;
    const int c0  = (kt & 7) * 32;
    const _Float16* srcA =
        hbuf + ((size_t)(t + tap) * B_ + b) * H_ + c0;
    Frag a;
    a.h[0] = *reinterpret_cast<const f16x8*>(srcA + lh * 8);
    a.h[1] = *reinterpret_cast<const f16x8*>(srcA + 16 + lh * 8);
#pragma unroll
    for (int j = 0; j < 4; ++j) {
      Frag bb;
      bb.h[0] = *reinterpret_cast<const f16x8*>(colB[j] + kt * 32);
      bb.h[1] = *reinterpret_cast<const f16x8*>(colB[j] + kt * 32 + 16);
      acc[j] = __builtin_amdgcn_wmma_f32_16x16x32_f16(false, a.v, false, bb.v,
                                                      (short)0, acc[j],
                                                      false, false);
    }
  }

#pragma unroll
  for (int j = 0; j < 4; ++j) {
#pragma unroll
    for (int r = 0; r < 8; ++r) {
      const int orow = mtile * 16 + r + 8 * lh;    // C/D layout row
      const float z = acc[j][r];
      out[(size_t)orow * 256 + n0 + j * 16 + lm] = z * sigf(z);
    }
  }
}

// ---------------------------------------------------------------------------
// Launch
// ---------------------------------------------------------------------------
extern "C" void kernel_launch(void* const* d_in, const int* in_sizes, int n_in,
                              void* d_out, int out_size, void* d_ws,
                              size_t ws_size, hipStream_t stream) {
  (void)in_sizes; (void)n_in; (void)out_size; (void)ws_size;

  const float* x    = (const float*)d_in[0];  // [64][1024][64]
  const float* W    = (const float*)d_in[1];  // [64][1024]
  const float* U    = (const float*)d_in[2];  // [256][1024]
  const float* bias = (const float*)d_in[3];  // [1024]
  const float* Wc   = (const float*)d_in[4];  // [5][256][256]
  const float* bc   = (const float*)d_in[5];  // [256]
  float* out = (float*)d_out;                 // [64][1020][256]

  char* ws = (char*)d_ws;
  _Float16* WcatT = (_Float16*)(ws);                 //   655,360 B
  _Float16* WcT   = (_Float16*)(ws + 655360);        //   655,360 B
  _Float16* hbuf  = (_Float16*)(ws + 1310720);       // 33,554,432 B

  hipLaunchKernelGGL(pack_wcat, dim3(1024), dim3(256), 0, stream, W, U, WcatT);
  hipLaunchKernelGGL(pack_wc,   dim3(256),  dim3(256), 0, stream, Wc, WcT);
  // 16 workgroups = one cluster of 16 WGPs; 4 waves each.
  hipLaunchKernelGGL(lstm_recur, dim3(16), dim3(128), 0, stream,
                     x, bias, WcatT, hbuf);
  // 4080 M tiles; 4 waves x 64 cols cover all 256 filters.
  hipLaunchKernelGGL(conv_silu, dim3(4080), dim3(128), 0, stream,
                     hbuf, WcT, bc, out);
}
